// Network_2121713845020
// MI455X (gfx1250) — matrix operand order repaired and checked by reading, not compiled, not run on hardware
//
#include <hip/hip_runtime.h>
#include <hip/hip_bf16.h>
#include <stdint.h>

// Problem dims (from reference)
#define N_TOK 16384
#define DIM   1024
#define HID   4096
#define NEXP  8
#define OUTD  4096

typedef __attribute__((ext_vector_type(16))) __bf16          v16bf;
typedef __attribute__((ext_vector_type(8)))  float           v8f;
typedef __attribute__((ext_vector_type(8)))  unsigned short  u16x8;
typedef __attribute__((ext_vector_type(4)))  unsigned short  u16x4;
typedef __attribute__((ext_vector_type(4)))  int             v4i;

struct Frag16 { u16x8 lo, hi; };

__device__ __forceinline__ unsigned short f2bf(float f) {
  unsigned u = __float_as_uint(f);
  u += 0x7FFFu + ((u >> 16) & 1u);         // round-to-nearest-even
  return (unsigned short)(u >> 16);
}
__device__ __forceinline__ float bf2f(unsigned short h) {
  return __uint_as_float(((unsigned)h) << 16);
}

// ---- gfx1250 async global->LDS copy (ASYNCcnt) with graceful fallback ----
#if defined(__HIP_DEVICE_COMPILE__) && __has_builtin(__builtin_amdgcn_global_load_async_to_lds_b128)
#define HAVE_ASYNC_LDS 1
#else
#define HAVE_ASYNC_LDS 0
#endif

#if HAVE_ASYNC_LDS
#if __has_builtin(__builtin_amdgcn_s_wait_asynccnt)
#define ASYNC_WAIT() __builtin_amdgcn_s_wait_asynccnt(0)
#else
#define ASYNC_WAIT() asm volatile("s_wait_asynccnt 0x0" ::: "memory")
#endif
// prototype (from clang diagnostic): (V4i AS1* src, V4i AS3* dst, imm offset, imm cpol)
#define ASYNC_CP16(dst, src)                                               \
  __builtin_amdgcn_global_load_async_to_lds_b128(                          \
      (__attribute__((address_space(1))) v4i*)(src),                       \
      (__attribute__((address_space(3))) v4i*)(dst), 0, 0)
#else
#define ASYNC_WAIT() ((void)0)
#define ASYNC_CP16(dst, src) (*(u16x8*)(dst) = *(const u16x8*)(src))
#endif

// ---------------- Tiled bf16 WMMA GEMM core ----------------
// Block tile 128x128, K-step 32, 8 wave32 waves: wave = 32(M) x 64(N) = 2x4 WMMA tiles.
// A: bf16 row-major [M][K] (optional row gather). B: bf16 pre-transposed [N][K].
// Both tiles are 128 rows x 64B, staged to double-buffered LDS via async b128 copies.
constexpr int BM = 128, BN = 128, BK = 32;
constexpr int LDA_S = BK + 8;   // padded pitch (halves): 80B rows stay 16B-aligned
constexpr int LDB_S = BK + 8;

enum { EPI_BF16_BIAS = 0, EPI_BF16_RELU_BIAS = 1, EPI_SCATTER = 2, EPI_F32 = 3 };

template <bool A_GATHER, int EPI>
__global__ __launch_bounds__(256)
void gemm_wmma(const unsigned short* __restrict__ A, int lda,
               const unsigned short* __restrict__ Bt, int ldb,   // [N][K], ldb == K
               const float* __restrict__ bias,
               void* __restrict__ Cptr, int ldc,
               int K, int Mmax,
               const int*   __restrict__ countPtr,   // nullptr => dense (Mmax rows)
               const int*   __restrict__ gidx,       // gather/scatter token list
               const float* __restrict__ gwts)       // scatter weights
{
  __shared__ unsigned short As[2][BM * LDA_S];
  __shared__ unsigned short Bs[2][BN * LDB_S];

  const int rows = countPtr ? *countPtr : Mmax;
  const int m0 = blockIdx.y * BM;
  if (m0 >= rows) return;                       // uniform early-exit (routed experts)
  const int n0 = blockIdx.x * BN;

  const int tid  = threadIdx.x;
  const int lane = tid & 31;
  const int wave = tid >> 5;
  const int wm = (wave & 3) * 32;               // wave M offset in block
  const int wn = (wave >> 2) * 64;              // wave N offset in block

  v8f acc[2][4];
#pragma unroll
  for (int i = 0; i < 2; ++i)
#pragma unroll
    for (int j = 0; j < 4; ++j) acc[i][j] = (v8f){0.f,0.f,0.f,0.f,0.f,0.f,0.f,0.f};

  // Copy map: each of 256 threads owns one (row, 16-half chunk): 128 rows x 2 chunks.
  const int cp_row = tid >> 1;            // 0..127
  const int cp_off = (tid & 1) * 16;      // halves: 0 or 16
  int aRowIdx;
  {
    const int gm = m0 + cp_row;
    if (A_GATHER) aRowIdx = (gm < rows) ? gidx[gm] : 0;  // never chase stale indices
    else          aRowIdx = gm;                          // buffer always has Mmax rows
  }
  const unsigned short* aRow = A  + (size_t)aRowIdx * lda;
  const unsigned short* bRow = Bt + (size_t)(n0 + cp_row) * ldb;

  auto stage_tile = [&](int bufIdx, int kt) {
    const unsigned short* asrc = aRow + kt + cp_off;
    const unsigned short* bsrc = bRow + kt + cp_off;
    unsigned short* adst = &As[bufIdx][cp_row * LDA_S + cp_off];
    unsigned short* bdst = &Bs[bufIdx][cp_row * LDB_S + cp_off];
    ASYNC_CP16(adst,     asrc);
    ASYNC_CP16(adst + 8, asrc + 8);
    ASYNC_CP16(bdst,     bsrc);
    ASYNC_CP16(bdst + 8, bsrc + 8);
  };

  const int hh  = lane >> 4;        // half-wave select (K striping per ISA layout)
  const int l16 = lane & 15;

  int buf = 0;
  stage_tile(0, 0);
  ASYNC_WAIT();
  __syncthreads();

  for (int kt = 0; kt < K; kt += BK) {
    if (kt + BK < K) stage_tile(buf ^ 1, kt + BK);   // prefetch next tile (overlapped)

    // ---- fragment loads (CDNA5 16-bit A/B VGPR striping) ----
    v16bf afrag[2];
#pragma unroll
    for (int i = 0; i < 2; ++i) {
      const unsigned short* base = &As[buf][(wm + i * 16 + l16) * LDA_S + hh * 8];
      Frag16 f; f.lo = *(const u16x8*)base; f.hi = *(const u16x8*)(base + 16);
      afrag[i] = __builtin_bit_cast(v16bf, f);
    }
    v16bf bfrag[4];
#pragma unroll
    for (int j = 0; j < 4; ++j) {
      const unsigned short* base = &Bs[buf][(wn + j * 16 + l16) * LDB_S + hh * 8];
      Frag16 f; f.lo = *(const u16x8*)base; f.hi = *(const u16x8*)(base + 16);
      bfrag[j] = __builtin_bit_cast(v16bf, f);
    }
#pragma unroll
    for (int i = 0; i < 2; ++i)
#pragma unroll
      for (int j = 0; j < 4; ++j)
        acc[i][j] = __builtin_amdgcn_wmma_f32_16x16x32_bf16(
            false, afrag[i], false, bfrag[j], (short)0, acc[i][j], false, false);

    ASYNC_WAIT();          // prefetch issued above must land before next iteration
    __syncthreads();
    buf ^= 1;
  }

  // ---- epilogue (C layout: VGPR r -> row wm+i*16+r+hh*8, col = lane&15) ----
#pragma unroll
  for (int i = 0; i < 2; ++i) {
#pragma unroll
    for (int j = 0; j < 4; ++j) {
      const int ncol = n0 + wn + j * 16 + l16;
      const float bv = (EPI == EPI_F32) ? 0.f : bias[ncol];
#pragma unroll
      for (int r = 0; r < 8; ++r) {
        const int mrow = m0 + wm + i * 16 + r + hh * 8;
        if (mrow >= rows) continue;
        const float v = acc[i][j][r] + bv;
        if (EPI == EPI_BF16_BIAS) {
          ((unsigned short*)Cptr)[(size_t)mrow * ldc + ncol] = f2bf(v);
        } else if (EPI == EPI_BF16_RELU_BIAS) {
          ((unsigned short*)Cptr)[(size_t)mrow * ldc + ncol] = f2bf(fmaxf(v, 0.f));
        } else if (EPI == EPI_F32) {
          ((float*)Cptr)[(size_t)mrow * ldc + ncol] = v;
        } else { // EPI_SCATTER: moe[token] += w * (acc + bias)
          const int   token = gidx[mrow];
          const float w     = gwts[mrow];
          float* dst = (float*)Cptr + (size_t)token * ldc + ncol;
          *dst += w * v;   // tokens unique per expert list; experts serialized on stream
        }
      }
    }
  }
}

// ---------------- fp32 -> bf16 dense convert ----------------
__global__ void cvt_dense(const float* __restrict__ src, unsigned short* __restrict__ dst,
                          long long n)
{
  const long long stride = 4LL * gridDim.x * blockDim.x;
  for (long long i = 4LL * ((long long)blockIdx.x * blockDim.x + threadIdx.x);
       i < n; i += stride) {
    const float4 v = *(const float4*)(src + i);
    u16x4 h; h.x = f2bf(v.x); h.y = f2bf(v.y); h.z = f2bf(v.z); h.w = f2bf(v.w);
    *(u16x4*)(dst + i) = h;
  }
}

// ---------------- fp32 [R][C] -> bf16 transposed [C][R] (per-z matrix) ----------------
__global__ __launch_bounds__(256)
void cvt_transpose(const float* __restrict__ src, unsigned short* __restrict__ dst,
                   int R, int C, size_t srcStride, size_t dstStride)
{
  __shared__ unsigned short t[32][33];
  const float* s = src + (size_t)blockIdx.z * srcStride;
  unsigned short* d = dst + (size_t)blockIdx.z * dstStride;
  const int c  = threadIdx.x & 31;
  const int r0 = threadIdx.x >> 5;
  const int rb = blockIdx.y * 32, cb = blockIdx.x * 32;
#pragma unroll
  for (int i = 0; i < 4; ++i) {
    const int r = r0 + i * 8;
    t[r][c] = f2bf(s[(size_t)(rb + r) * C + cb + c]);
  }
  __syncthreads();
#pragma unroll
  for (int i = 0; i < 4; ++i) {
    const int r = r0 + i * 8;
    d[(size_t)(cb + r) * R + rb + c] = t[c][r];
  }
}

// ---------------- gating: logits -> top-2 -> renormalized weights + routing lists ----------------
__global__ __launch_bounds__(256)
void gating_kernel(const unsigned short* __restrict__ hbf,
                   const float* __restrict__ Wg,
                   int* __restrict__ counts,
                   int* __restrict__ list,
                   float* __restrict__ wts)
{
  const int token = blockIdx.x * 8 + (threadIdx.x >> 5);
  const int lane  = threadIdx.x & 31;
  float acc[NEXP];
#pragma unroll
  for (int e = 0; e < NEXP; ++e) acc[e] = 0.f;

  const unsigned short* hrow = hbf + (size_t)token * DIM;
  for (int d = lane; d < DIM; d += 32) {
    const float hv = bf2f(hrow[d]);
    const float4 w0 = *(const float4*)(Wg + (size_t)d * NEXP);
    const float4 w1 = *(const float4*)(Wg + (size_t)d * NEXP + 4);
    acc[0] += hv * w0.x; acc[1] += hv * w0.y; acc[2] += hv * w0.z; acc[3] += hv * w0.w;
    acc[4] += hv * w1.x; acc[5] += hv * w1.y; acc[6] += hv * w1.z; acc[7] += hv * w1.w;
  }
#pragma unroll
  for (int off = 16; off; off >>= 1)
#pragma unroll
    for (int e = 0; e < NEXP; ++e) acc[e] += __shfl_xor(acc[e], off, 32);

  if (lane == 0) {
    int i0 = 0;
#pragma unroll
    for (int e = 1; e < NEXP; ++e) if (acc[e] > acc[i0]) i0 = e;
    int i1 = (i0 == 0) ? 1 : 0;
#pragma unroll
    for (int e = 0; e < NEXP; ++e) if (e != i0 && acc[e] > acc[i1]) i1 = e;
    // softmax restricted to top-2 == renormalized top-2 softmax
    const float t  = __expf(acc[i1] - acc[i0]);
    const float w0 = 1.f / (1.f + t);
    const float w1 = t / (1.f + t);
    int p0 = atomicAdd(&counts[i0], 1);
    list[i0 * N_TOK + p0] = token; wts[i0 * N_TOK + p0] = w0;
    int p1 = atomicAdd(&counts[i1], 1);
    list[i1 * N_TOK + p1] = token; wts[i1 * N_TOK + p1] = w1;
  }
}

__global__ void init_kernel(float* __restrict__ moe, int* __restrict__ counts, long long n)
{
  long long i = (long long)blockIdx.x * blockDim.x + threadIdx.x;
  const long long stride = (long long)gridDim.x * blockDim.x;
  for (; i < n; i += stride) moe[i] = 0.f;
  if (blockIdx.x == 0 && threadIdx.x < NEXP) counts[threadIdx.x] = 0;
}

// ---------------- driver ----------------
extern "C" void kernel_launch(void* const* d_in, const int* in_sizes, int n_in,
                              void* d_out, int out_size, void* d_ws, size_t ws_size,
                              hipStream_t stream)
{
  const float* x      = (const float*)d_in[0];
  const float* W_in   = (const float*)d_in[1];
  const float* b_in   = (const float*)d_in[2];
  const float* W_gate = (const float*)d_in[3];
  const float* W1     = (const float*)d_in[4];
  const float* b1     = (const float*)d_in[5];
  const float* W2     = (const float*)d_in[6];
  const float* b2     = (const float*)d_in[7];
  const float* W_head = (const float*)d_in[8];

  typedef unsigned short u16;
  char* ws = (char*)d_ws;
  size_t off = 0;
  u16*   xbf   = (u16*)(ws + off);  off += (size_t)N_TOK * DIM * 2;        //  33.5 MB
  u16*   hbf   = (u16*)(ws + off);  off += (size_t)N_TOK * DIM * 2;        //  33.5 MB
  u16*   hebf  = (u16*)(ws + off);  off += (size_t)N_TOK * HID * 2;        // 134.2 MB
  float* moe   = (float*)(ws + off); off += (size_t)N_TOK * DIM * 4;       //  67.1 MB
  u16*   moebf = (u16*)(ws + off);  off += (size_t)N_TOK * DIM * 2;        //  33.5 MB
  u16*   Wint  = (u16*)(ws + off);  off += (size_t)DIM * DIM * 2;          //   2.1 MB
  u16*   W1t   = (u16*)(ws + off);  off += (size_t)NEXP * DIM * HID * 2;   //  67.1 MB
  u16*   W2t   = (u16*)(ws + off);  off += (size_t)NEXP * HID * DIM * 2;   //  67.1 MB
  u16*   Wht   = (u16*)(ws + off);  off += (size_t)DIM * OUTD * 2;         //   8.4 MB
  int*   list  = (int*)(ws + off);  off += (size_t)NEXP * N_TOK * 4;
  float* wts   = (float*)(ws + off); off += (size_t)NEXP * N_TOK * 4;
  int*   cnts  = (int*)(ws + off);  off += 256;

  const dim3 blk(256);

  // zero moe accumulator + routing counters (deterministic per call)
  init_kernel<<<1024, blk, 0, stream>>>(moe, cnts, (long long)N_TOK * DIM);

  // one-time (per call) bf16 conversions; weights also transposed to [N][K]
  cvt_dense<<<1024, blk, 0, stream>>>(x, xbf, (long long)N_TOK * DIM);
  cvt_transpose<<<dim3(DIM / 32, DIM / 32, 1), blk, 0, stream>>>(
      W_in, Wint, DIM, DIM, 0, 0);
  cvt_transpose<<<dim3(HID / 32, DIM / 32, NEXP), blk, 0, stream>>>(
      W1, W1t, DIM, HID, (size_t)DIM * HID, (size_t)HID * DIM);
  cvt_transpose<<<dim3(DIM / 32, HID / 32, NEXP), blk, 0, stream>>>(
      W2, W2t, HID, DIM, (size_t)HID * DIM, (size_t)DIM * HID);
  cvt_transpose<<<dim3(OUTD / 32, DIM / 32, 1), blk, 0, stream>>>(
      W_head, Wht, DIM, OUTD, 0, 0);

  // h = x @ W_in + b_in  ->  bf16
  gemm_wmma<false, EPI_BF16_BIAS>
      <<<dim3(DIM / BN, N_TOK / BM), blk, 0, stream>>>(
          xbf, DIM, Wint, DIM, b_in, hbf, DIM, DIM, N_TOK, nullptr, nullptr, nullptr);

  // gating: top-2 routing lists + renormalized weights
  gating_kernel<<<N_TOK / 8, blk, 0, stream>>>(hbf, W_gate, cnts, list, wts);

  // routed expert FFNs (only top-2 tokens per expert; ~4x fewer FLOPs than dense)
  for (int e = 0; e < NEXP; ++e) {
    gemm_wmma<true, EPI_BF16_RELU_BIAS>
        <<<dim3(HID / BN, N_TOK / BM), blk, 0, stream>>>(
            hbf, DIM, W1t + (size_t)e * DIM * HID, DIM, b1 + (size_t)e * HID,
            hebf, HID, DIM, N_TOK, cnts + e, list + (size_t)e * N_TOK, nullptr);
    gemm_wmma<false, EPI_SCATTER>
        <<<dim3(DIM / BN, N_TOK / BM), blk, 0, stream>>>(
            hebf, HID, W2t + (size_t)e * HID * DIM, HID, b2 + (size_t)e * DIM,
            moe, DIM, HID, N_TOK, cnts + e, list + (size_t)e * N_TOK,
            wts + (size_t)e * N_TOK);
  }

  // out = moe @ W_head (fp32 moe -> bf16, then clean bf16 GEMM path)
  cvt_dense<<<1024, blk, 0, stream>>>(moe, moebf, (long long)N_TOK * DIM);
  gemm_wmma<false, EPI_F32>
      <<<dim3(OUTD / BN, N_TOK / BM), blk, 0, stream>>>(
          moebf, DIM, Wht, DIM, nullptr, d_out, OUTD, DIM, N_TOK,
          nullptr, nullptr, nullptr);
}
